// RNN_51659866637107
// MI455X (gfx1250) — matrix operand order repaired and checked by reading, not compiled, not run on hardware
//
#include <hip/hip_runtime.h>

#define B_   4
#define S_   2048
#define DIN  2048
#define DST  2048
#define DOUT 2048
#define NH   16
#define HH   128

typedef __attribute__((ext_vector_type(16))) __bf16 v16bf;
typedef __attribute__((ext_vector_type(8)))  float  v8f;

union Frag16 { v16bf v; uint4 u[2]; };
static_assert(sizeof(v16bf) == 32, "v16bf size");

// ---- optional CDNA5 async global->LDS staging path ------------------------
#if defined(__has_builtin)
#if __has_builtin(__builtin_amdgcn_global_load_async_to_lds_b128) && \
    __has_builtin(__builtin_amdgcn_s_wait_asynccnt)
#define USE_ASYNC_LDS 1
#endif
#endif
#ifndef USE_ASYNC_LDS
#define USE_ASYNC_LDS 0
#endif

#if USE_ASYNC_LDS
// Builtin signature (from hipcc diagnostic): param0 = AS1 pointer to
// 'int __attribute__((vector_size(16)))'; param1 = LDS-side pointer.
typedef int v4i_vs __attribute__((vector_size(16)));
typedef __attribute__((address_space(1))) v4i_vs gv4i_t;
typedef __attribute__((address_space(3))) v4i_vs lv4i_t;
#endif

__device__ __forceinline__ unsigned short f32_to_bf16(float f) {
  unsigned int u = __float_as_uint(f);
  u += 0x7fffu + ((u >> 16) & 1u);   // round-to-nearest-even
  return (unsigned short)(u >> 16);
}

// ---------------- fp32 -> bf16 bulk convert (4 elems/thread) ----------------
__global__ void __launch_bounds__(256)
cvt_f32_bf16(const float* __restrict__ in, unsigned short* __restrict__ out, int n) {
  int i = (blockIdx.x * 256 + threadIdx.x) * 4;
  if (i < n) {
    float4 v = *(const float4*)(in + i);
    unsigned int lo = (unsigned int)f32_to_bf16(v.x) | ((unsigned int)f32_to_bf16(v.y) << 16);
    unsigned int hi = (unsigned int)f32_to_bf16(v.z) | ((unsigned int)f32_to_bf16(v.w) << 16);
    *(uint2*)(out + i) = make_uint2(lo, hi);
  }
}

// ------------- transpose + convert: in[K][N] f32 -> out[N][K] bf16 ----------
__global__ void __launch_bounds__(256)
tconv_2048(const float* __restrict__ in, unsigned short* __restrict__ out) {
  __shared__ float tile[32][33];
  const int n0 = blockIdx.x * 32, k0 = blockIdx.y * 32;
  const int tx = threadIdx.x & 31, ty = threadIdx.x >> 5;
  for (int r = ty; r < 32; r += 8)
    tile[r][tx] = in[(size_t)(k0 + r) * 2048 + n0 + tx];
  __syncthreads();
  for (int r = ty; r < 32; r += 8)
    out[(size_t)(n0 + r) * 2048 + k0 + tx] = f32_to_bf16(tile[tx][r]);
}

// ---------------- bf16 GEMM: C[M][N] = A[M][K] * Bt[N][K]^T (+bias) ---------
// 256 threads = 8 waves as 2(M) x 4(N). WG tile 64(M) x 256(N), K-step 32.
// Each wave: 32x64 = 8 wmma accumulators; A frags reused x4, B frags x2.
#define ATILE (64 * 40)     // ushorts per A LDS buffer (row stride 40 = pad)
#define BTILE (256 * 40)    // ushorts per B LDS buffer

__device__ __forceinline__ void async_cp16(const unsigned short* g, unsigned short* l) {
#if USE_ASYNC_LDS
  __builtin_amdgcn_global_load_async_to_lds_b128((gv4i_t*)g, (lv4i_t*)l, 0, 0);
#else
  (void)g; (void)l;
#endif
}

__device__ __forceinline__ void gemm_compute_step(
    const unsigned short* bufA, const unsigned short* bufB,
    int wm, int wn, int mlo, int ksel, v8f acc[2][4])
{
  Frag16 a[2];
#pragma unroll
  for (int i = 0; i < 2; ++i) {
    const unsigned short* ap = bufA + (wm * 32 + i * 16 + mlo) * 40;
    a[i].u[0] = *(const uint4*)(ap + ksel * 8);        // K = 8*ksel .. +7
    a[i].u[1] = *(const uint4*)(ap + 16 + ksel * 8);   // K = 16+8*ksel .. +7
  }
#pragma unroll
  for (int j = 0; j < 4; ++j) {
    Frag16 b;
    const unsigned short* bp = bufB + (wn * 64 + j * 16 + mlo) * 40 + ksel * 8;
    b.u[0] = *(const uint4*)(bp);                      // K = 8*ksel .. +7
    b.u[1] = *(const uint4*)(bp + 16);                 // K = 16+8*ksel .. +7
    acc[0][j] = __builtin_amdgcn_wmma_f32_16x16x32_bf16(
        false, a[0].v, false, b.v, (short)0, acc[0][j], false, false);
    acc[1][j] = __builtin_amdgcn_wmma_f32_16x16x32_bf16(
        false, a[1].v, false, b.v, (short)0, acc[1][j], false, false);
  }
}

__global__ void __launch_bounds__(256)
gemm_bf16_tn(const unsigned short* __restrict__ A,
             const unsigned short* __restrict__ Bt,
             const float* __restrict__ bias,
             float* __restrict__ C, int M, int N, int K)
{
  __shared__ __align__(16) unsigned short lsA[2 * ATILE];  // 10 KB
  __shared__ __align__(16) unsigned short lsB[2 * BTILE];  // 40 KB
  const int tid  = threadIdx.x;
  const int wave = tid >> 5, lane = tid & 31;
  const int wm   = wave >> 2, wn = wave & 3;
  const int mlo  = lane & 15, ksel = lane >> 4;
  const int m0   = blockIdx.y * 64, n0 = blockIdx.x * 256;
  const int ksteps = K >> 5;
  const int ra = tid >> 2, qa = tid & 3;   // A staging: 64 rows x 4 quarters

  v8f acc[2][4] = {};

#if USE_ASYNC_LDS
  // ---- double-buffered async pipeline (5 async instrs per thread per tile)
  {
    async_cp16(A + (size_t)(m0 + ra) * K + qa * 8, lsA + ra * 40 + qa * 8);
#pragma unroll
    for (int c = 0; c < 4; ++c)
      async_cp16(Bt + (size_t)(n0 + tid) * K + c * 8, lsB + tid * 40 + c * 8);
  }
  for (int kt = 0; kt < ksteps; ++kt) {
    __builtin_amdgcn_s_wait_asynccnt(0);   // this wave's tile kt landed in LDS
    __syncthreads();                       // every wave's tile kt landed
    const int cur = kt & 1, nxt = cur ^ 1;
    if (kt + 1 < ksteps) {
      const int k1 = (kt + 1) << 5;
      async_cp16(A + (size_t)(m0 + ra) * K + k1 + qa * 8,
                 lsA + nxt * ATILE + ra * 40 + qa * 8);
#pragma unroll
      for (int c = 0; c < 4; ++c)
        async_cp16(Bt + (size_t)(n0 + tid) * K + k1 + c * 8,
                   lsB + nxt * BTILE + tid * 40 + c * 8);
    }
    gemm_compute_step(lsA + cur * ATILE, lsB + cur * BTILE, wm, wn, mlo, ksel, acc);
  }
#else
  // ---- synchronous staging fallback (single buffer)
  for (int kt = 0; kt < ksteps; ++kt) {
    const int k0 = kt << 5;
    uint4 va = *(const uint4*)(A + (size_t)(m0 + ra) * K + k0 + qa * 8);
    uint4 vb[4];
#pragma unroll
    for (int c = 0; c < 4; ++c)
      vb[c] = *(const uint4*)(Bt + (size_t)(n0 + tid) * K + k0 + c * 8);
    *(uint4*)(lsA + ra * 40 + qa * 8) = va;
#pragma unroll
    for (int c = 0; c < 4; ++c)
      *(uint4*)(lsB + tid * 40 + c * 8) = vb[c];
    __syncthreads();
    gemm_compute_step(lsA, lsB, wm, wn, mlo, ksel, acc);
    __syncthreads();
  }
#endif

  // ---- epilogue: lane holds col n0+wn*64+j*16+mlo, rows +ksel*8+r
#pragma unroll
  for (int i = 0; i < 2; ++i) {
#pragma unroll
    for (int j = 0; j < 4; ++j) {
      const int col  = n0 + wn * 64 + j * 16 + mlo;
      const float ba = bias ? bias[col] : 0.0f;
      const int rowb = m0 + wm * 32 + i * 16 + ksel * 8;
#pragma unroll
      for (int r = 0; r < 8; ++r)
        C[(size_t)(rowb + r) * N + col] = acc[i][j][r] + ba;
    }
  }
}

// ---------------- sequential recurrence, one head per workgroup -------------
__global__ void __launch_bounds__(512)
rnn_scan(const float* __restrict__ xp,        // [B,S,DST] f32
         const float* __restrict__ state_w,   // [NH,HH,HH] f32
         const float* __restrict__ h0,        // [B,DST] f32
         unsigned short* __restrict__ hs,     // [B,S,DST] bf16 out
         float* __restrict__ out_state)       // [B,DST] f32 out
{
  extern __shared__ float smem[];
  float* Wt = smem;               // [128][132] transposed head weights
  float* hb = smem + 128 * 132;   // [4][128] hidden state
  const int n   = blockIdx.x;
  const int tid = threadIdx.x;
  const int b   = tid >> 7, k = tid & 127;

  const float* Wn = state_w + (size_t)n * HH * HH;
  for (int idx = tid; idx < HH * HH; idx += 512) {
    int i = idx >> 7, kk = idx & 127;
    Wt[kk * 132 + i] = Wn[idx];
  }
  hb[b * 128 + k] = h0[(size_t)b * DST + n * HH + k];
  __syncthreads();

  const float* wrow = Wt + k * 132;
  const float* hrow = hb + b * 128;
  const size_t gcol = (size_t)n * HH + k;

  for (int t = 0; t < S_; ++t) {
    float sum = xp[((size_t)b * S_ + t) * DST + gcol];
#pragma unroll 8
    for (int i = 0; i < 128; i += 4) {
      float4 w = *(const float4*)(wrow + i);
      float4 h = *(const float4*)(hrow + i);
      sum += w.x * h.x + w.y * h.y + w.z * h.z + w.w * h.w;
    }
    float nh = tanhf(sum);
    __syncthreads();
    hb[b * 128 + k] = nh;
    hs[((size_t)b * S_ + t) * DST + gcol] = f32_to_bf16(nh);
    if (t == S_ - 1) out_state[(size_t)b * DST + gcol] = nh;
    __syncthreads();
  }
}

// ---------------------------------------------------------------------------
extern "C" void kernel_launch(void* const* d_in, const int* in_sizes, int n_in,
                              void* d_out, int out_size, void* d_ws, size_t ws_size,
                              hipStream_t stream) {
  const float* x    = (const float*)d_in[0];
  const float* h0   = (const float*)d_in[1];
  const float* Wi   = (const float*)d_in[2];
  const float* bias = (const float*)d_in[3];
  const float* sw   = (const float*)d_in[4];
  const float* Wo   = (const float*)d_in[5];

  float* out       = (float*)d_out;
  float* out_state = out + (size_t)B_ * S_ * DOUT;

  char* ws = (char*)d_ws;
  unsigned short* xbf = (unsigned short*)(ws);                      // 33.5 MB
  unsigned short* WiT = (unsigned short*)(ws + (size_t)33554432);   //  8.4 MB
  unsigned short* WoT = (unsigned short*)(ws + (size_t)41943040);   //  8.4 MB
  float*          xp  = (float*)        (ws + (size_t)50331648);    // 67.1 MB
  unsigned short* hsb = (unsigned short*)(ws + (size_t)117440512);  // 33.5 MB

  const int M = B_ * S_;            // 8192
  const int nx = B_ * S_ * DIN;     // 16.7M

  cvt_f32_bf16<<<nx / 4 / 256, 256, 0, stream>>>(x, xbf, nx);
  tconv_2048<<<dim3(64, 64), 256, 0, stream>>>(Wi, WiT);
  tconv_2048<<<dim3(64, 64), 256, 0, stream>>>(Wo, WoT);

  gemm_bf16_tn<<<dim3(DST / 256, M / 64), 256, 0, stream>>>(
      xbf, WiT, bias, xp, M, DST, DIN);

  const size_t scan_lds = (size_t)(128 * 132 + 4 * 128) * sizeof(float); // 69632 B
  rnn_scan<<<NH, 512, scan_lds, stream>>>(xp, sw, h0, hsb, out_state);

  gemm_bf16_tn<<<dim3(DOUT / 256, M / 64), 256, 0, stream>>>(
      hsb, WoT, nullptr, out, M, DOUT, DST);
}
